// MambaDualMemController_24051816857856
// MI455X (gfx1250) — matrix operand order repaired and checked by reading, not compiled, not run on hardware
//
#include <hip/hip_runtime.h>
#include <hip/hip_bf16.h>

typedef _Float16 half16 __attribute__((ext_vector_type(16)));
typedef float    float8 __attribute__((ext_vector_type(8)));

#define CB 8
#define CT 2048
#define CD 512
#define CDI 1024
#define CL 4
#define CV 1024
#define CBT (CB * CT)

__device__ __forceinline__ float sigmoidf_(float x) { return 1.0f / (1.0f + __expf(-x)); }
__device__ __forceinline__ float softplus_(float x) { return (x > 20.0f) ? x : log1pf(__expf(x)); }

// ---------------------------------------------------------------------------
// Embedding gather: h[b,t,:] = embed[tok[b,t],:]
// ---------------------------------------------------------------------------
__global__ void embed_kernel(const int* __restrict__ tokens,
                             const float* __restrict__ embed,
                             float* __restrict__ h, int total)
{
    int idx = blockIdx.x * blockDim.x + threadIdx.x;
    if (idx >= total) return;
    int d  = idx & (CD - 1);
    int bt = idx >> 9;
    h[idx] = embed[tokens[bt] * CD + d];
}

// ---------------------------------------------------------------------------
// LayerNorm over D=512, one wave per row (wave32: 16 elems per lane)
// ---------------------------------------------------------------------------
__global__ void __launch_bounds__(128) ln_kernel(const float* __restrict__ x,
                                                 const float* __restrict__ w,
                                                 const float* __restrict__ b,
                                                 float* __restrict__ out, int nrows)
{
    int wid = threadIdx.x >> 5, lane = threadIdx.x & 31;
    int row = blockIdx.x * 4 + wid;
    if (row >= nrows) return;
    const float* xr = x + (size_t)row * CD;
    float vals[16];
    float s = 0.0f;
#pragma unroll
    for (int i = 0; i < 16; ++i) { vals[i] = xr[lane + i * 32]; s += vals[i]; }
#pragma unroll
    for (int off = 16; off >= 1; off >>= 1) s += __shfl_xor(s, off, 32);
    float mean = s * (1.0f / (float)CD);
    float v = 0.0f;
#pragma unroll
    for (int i = 0; i < 16; ++i) { float d0 = vals[i] - mean; v += d0 * d0; }
#pragma unroll
    for (int off = 16; off >= 1; off >>= 1) v += __shfl_xor(v, off, 32);
    float inv = rsqrtf(v * (1.0f / (float)CD) + 1e-5f);
    float* orow = out + (size_t)row * CD;
#pragma unroll
    for (int i = 0; i < 16; ++i) {
        int c = lane + i * 32;
        orow[c] = (vals[i] - mean) * inv * w[c] + b[c];
    }
}

// ---------------------------------------------------------------------------
// WMMA GEMM: C[M,N] = act(A[M,K] @ B[K,N] + bias) (+C residual)
// f32 in/out, f16 operands, v_wmma_f32_16x16x32_f16.
// Block tile 64x64, BK=32, 256 threads = 8 waves, each wave does a 16x32 strip.
// LDS staging:
//   As row-major   64 rows x 32 f16, row stride 40 f16 (80 B: 16B-aligned rows,
//                  20-bank stride -> conflict-free b128 fragment loads)
//   Bs COLUMN-major 64 cols x 32 f16, col stride 40 f16 (same property), so a
//                  lane's B fragment is 16 contiguous f16 -> ds_load_b128 x2
// Requires M%64==0, N%64==0, K%32==0 (true for every call site here).
// ACT: 0=none, 1=softplus.  BIAS: add bias[n].  RES: C += result.
// ---------------------------------------------------------------------------
#define SA 40
#define SB 40

template <int ACT, bool BIAS, bool RES>
__global__ void __launch_bounds__(256)
gemm_wmma(const float* __restrict__ A, const float* __restrict__ B,
          const float* __restrict__ bias, float* __restrict__ C,
          int M, int N, int K, int lda, int ldb, int ldc)
{
    __shared__ _Float16 As[64 * SA];
    __shared__ _Float16 Bs[64 * SB];

    const int tid  = threadIdx.x;
    const int lane = tid & 31;
    const int wid  = tid >> 5;
    const int g    = lane >> 4;    // lane half
    const int ln   = lane & 15;
    const int mrow = wid >> 1;     // 0..3 -> 16-row strip
    const int npair = wid & 1;     // 0..1 -> 32-col strip (two 16-wide tiles)
    const int blockM = blockIdx.y * 64;
    const int blockN = blockIdx.x * 64;

    // Per-thread staging coordinates (float4 granularity)
    const int ar0 = tid >> 3;             // A: rows 0..31 (+32 on 2nd iter)
    const int ac4 = (tid & 7) * 4;        // A: col group 0..28
    const int br0 = tid >> 4;             // B: rows 0..15 (+16 on 2nd iter)
    const int bc4 = (tid & 15) * 4;       // B: col group 0..60

    float8 c0 = {};
    float8 c1 = {};

    for (int k0 = 0; k0 < K; k0 += 32) {
        // ---- Stage A (64x32) as f16, row-major, padded stride ----
#pragma unroll
        for (int i = 0; i < 2; ++i) {
            int r = ar0 + i * 32;
            float4 av = *(const float4*)(A + (size_t)(blockM + r) * lda + (k0 + ac4));
            _Float16* dst = &As[r * SA + ac4];
            dst[0] = (_Float16)av.x; dst[1] = (_Float16)av.y;
            dst[2] = (_Float16)av.z; dst[3] = (_Float16)av.w;
        }
        // ---- Stage B (32x64) as f16, COLUMN-major, padded stride ----
#pragma unroll
        for (int i = 0; i < 2; ++i) {
            int r = br0 + i * 16;
            float4 bv = *(const float4*)(B + (size_t)(k0 + r) * ldb + (blockN + bc4));
            Bs[(bc4 + 0) * SB + r] = (_Float16)bv.x;
            Bs[(bc4 + 1) * SB + r] = (_Float16)bv.y;
            Bs[(bc4 + 2) * SB + r] = (_Float16)bv.z;
            Bs[(bc4 + 3) * SB + r] = (_Float16)bv.w;
        }
        // Prefetch next K-tile into cache while WMMA runs on this one.
        if (k0 + 32 < K) {
            __builtin_prefetch(A + (size_t)(blockM + ar0) * lda + (k0 + 32 + ac4), 0, 1);
            __builtin_prefetch(B + (size_t)(k0 + 32 + br0) * ldb + (blockN + bc4), 0, 1);
        }
        __syncthreads();

        // A fragment: 16x32 f16: lane-half g holds K {16g..16g+15} (contiguous, matches B)
        half16 a;
        {
            const _Float16* ap = &As[(mrow * 16 + ln) * SA + g * 16];
#pragma unroll
            for (int j = 0; j < 8; ++j) { a[j] = ap[j]; a[8 + j] = ap[8 + j]; }
        }
        // B fragments: contiguous 16 f16 per lane in column-major Bs.
        half16 b0, b1;
        {
            const _Float16* bp0 = &Bs[(npair * 32 + ln) * SB + g * 16];
            const _Float16* bp1 = bp0 + 16 * SB;
#pragma unroll
            for (int j = 0; j < 16; ++j) { b0[j] = bp0[j]; b1[j] = bp1[j]; }
        }
        c0 = __builtin_amdgcn_wmma_f32_16x16x32_f16(false, a, false, b0, (short)0, c0, false, false);
        c1 = __builtin_amdgcn_wmma_f32_16x16x32_f16(false, a, false, b1, (short)0, c1, false, false);
        __syncthreads();
    }

    // Store: C/D layout -> VGPR i holds row (8*g + i), col = ln.
#pragma unroll
    for (int i = 0; i < 8; ++i) {
        int m  = blockM + mrow * 16 + g * 8 + i;
        int n0 = blockN + npair * 32 + ln;
        int n1 = n0 + 16;
        float v0 = c0[i], v1 = c1[i];
        if (BIAS) { v0 += bias[n0]; v1 += bias[n1]; }
        if (ACT == 1) { v0 = softplus_(v0); v1 = softplus_(v1); }
        size_t o0 = (size_t)m * ldc + n0;
        size_t o1 = (size_t)m * ldc + n1;
        if (RES) { v0 += C[o0]; v1 += C[o1]; }
        C[o0] = v0;
        C[o1] = v1;
    }
}

// ---------------------------------------------------------------------------
// Depthwise causal conv (DC=4) + SiLU: xz[:, :DI] -> xc
// ---------------------------------------------------------------------------
__global__ void conv_silu_kernel(const float* __restrict__ xz,
                                 const float* __restrict__ conv_w,
                                 const float* __restrict__ conv_b,
                                 float* __restrict__ xc, int total)
{
    int idx = blockIdx.x * blockDim.x + threadIdx.x;
    if (idx >= total) return;
    int d  = idx & (CDI - 1);
    int bt = idx >> 10;
    int t  = bt & (CT - 1);
    const float* w = conv_w + d * 4;
    float acc = conv_b[d];
#pragma unroll
    for (int k = 0; k < 4; ++k) {
        int tt = t + k - 3;
        if (tt >= 0) acc += xz[(size_t)(bt + (k - 3)) * (2 * CDI) + d] * w[k];
    }
    xc[idx] = acc * sigmoidf_(acc);
}

// ---------------------------------------------------------------------------
// Mamba selective scan. One thread per (b,d); 16-float state in VGPRs.
// B_t/C_t (proj cols 32..63) broadcast through LDS. Writes y in-place to xc:
//   y = (scan_y + D*x) * silu(z)
// ---------------------------------------------------------------------------
__global__ void __launch_bounds__(256) scan_kernel(const float* __restrict__ dt,
                                                   const float* __restrict__ proj,
                                                   const float* __restrict__ xz,
                                                   const float* __restrict__ A_log,
                                                   const float* __restrict__ Dskip,
                                                   float* __restrict__ xc)
{
    __shared__ float sBC[32];
    const int tid = threadIdx.x;
    const int b = blockIdx.x >> 2;
    const int d = ((blockIdx.x & 3) << 8) + tid;

    float Aexp[16], s[16];
#pragma unroll
    for (int i = 0; i < 16; ++i) { Aexp[i] = -__expf(A_log[d * 16 + i]); s[i] = 0.0f; }
    const float Dk = Dskip[d];
    const size_t base = (size_t)b * CT;

    for (int t = 0; t < CT; ++t) {
        size_t row = base + t;
        if (tid < 32) sBC[tid] = proj[row * 64 + 32 + tid];
        __syncthreads();
        float dtv = dt[row * CDI + d];
        float xv  = xc[row * CDI + d];
        float zv  = xz[row * (2 * CDI) + CDI + d];
        float y = 0.0f;
#pragma unroll
        for (int i = 0; i < 16; ++i) {
            float dA = __expf(dtv * Aexp[i]);
            s[i] = dA * s[i] + (dtv * xv) * sBC[i];
            y += s[i] * sBC[16 + i];
        }
        xc[row * CDI + d] = (y + Dk * xv) * (zv * sigmoidf_(zv));
        __syncthreads();
    }
}

// ---------------------------------------------------------------------------
// Memory controller: 8 persistent workgroups (one per batch), 512 threads.
// K/V state (4 heads x 64 slots x 128) lives in dynamic LDS (256 KB < 320 KB/WGP).
// h is rewritten in-place with the fused+LN output per time step.
// ---------------------------------------------------------------------------
__global__ void __launch_bounds__(512) memctrl_kernel(float* __restrict__ h,
                                                      const float* __restrict__ Wq,
                                                      const float* __restrict__ Wk,
                                                      const float* __restrict__ Wv,
                                                      const float* __restrict__ Wo,
                                                      const float* __restrict__ gW,
                                                      const float* __restrict__ gb,
                                                      const float* __restrict__ lnw,
                                                      const float* __restrict__ lnb)
{
    extern __shared__ float lds_kv[];       // [0,32768): K, [32768,65536): V
    float* Km = lds_kv;
    float* Vm = lds_kv + 32768;
    __shared__ float s_h[CD], s_q[CD], s_k[CD], s_v[CD], s_read[CD], s_mo[CD];
    __shared__ float s_attn[256], s_addr[256];
    __shared__ float s_red[16];
    __shared__ float s_sc[2];

    const int tid = threadIdx.x, lane = tid & 31, wid = tid >> 5;
    const int b = blockIdx.x;
    const float scale = 0.08838834764831845f;   // 1/sqrt(128)

    for (int i = tid; i < 32768; i += 512) { Km[i] = 0.0f; Vm[i] = 0.0f; }
    __syncthreads();

    for (int t = 0; t < CT; ++t) {
        size_t row = ((size_t)b * CT + t) * CD;
        s_h[tid] = h[row + tid];
        __syncthreads();

        // q / kw / vw GEMVs (coalesced weight reads, L2-resident)
        float aq = 0.0f, ak = 0.0f, av = 0.0f;
        for (int j = 0; j < CD; ++j) {
            float hv = s_h[j];
            aq += hv * Wq[j * CD + tid];
            ak += hv * Wk[j * CD + tid];
            av += hv * Wv[j * CD + tid];
        }
        s_q[tid] = aq; s_k[tid] = ak; s_v[tid] = av;
        __syncthreads();

        // attention + addressing logits: 256 dots of length 128 over LDS K
        if (tid < 256) {
            int hh = tid >> 6, ss = tid & 63;
            const float* Kr = &Km[(hh * 64 + ss) * 128];
            const float* qh = &s_q[hh * 128];
            const float* kh = &s_k[hh * 128];
            float la = 0.0f, lb = 0.0f;
            for (int j = 0; j < 128; ++j) { float kv = Kr[j]; la += qh[j] * kv; lb += kh[j] * kv; }
            s_attn[tid] = la * scale;
            s_addr[tid] = lb * scale;
        }
        __syncthreads();

        // softmax over 64 slots per head: waves 0..3 -> attn, 4..7 -> addr
        if (wid < 8) {
            float* arr = (wid < 4) ? s_attn : s_addr;
            int hh = wid & 3;
            float v0 = arr[hh * 64 + lane], v1 = arr[hh * 64 + 32 + lane];
            float m = fmaxf(v0, v1);
            for (int off = 16; off >= 1; off >>= 1) m = fmaxf(m, __shfl_xor(m, off, 32));
            float e0 = __expf(v0 - m), e1 = __expf(v1 - m);
            float sm = e0 + e1;
            for (int off = 16; off >= 1; off >>= 1) sm += __shfl_xor(sm, off, 32);
            float inv = 1.0f / sm;
            arr[hh * 64 + lane] = e0 * inv;
            arr[hh * 64 + 32 + lane] = e1 * inv;
        }
        __syncthreads();

        // read = attn @ V
        {
            int hh = tid >> 7, dh = tid & 127;
            float r = 0.0f;
            for (int ss = 0; ss < 64; ++ss) r += s_attn[hh * 64 + ss] * Vm[(hh * 64 + ss) * 128 + dh];
            s_read[tid] = r;
        }
        __syncthreads();

        // mem_out GEMV + surprise reduction
        float mo = 0.0f;
        for (int j = 0; j < CD; ++j) mo += s_read[j] * Wo[j * CD + tid];
        s_mo[tid] = mo;
        float df = s_h[tid] - mo;
        float sq = df * df;
        for (int off = 16; off >= 1; off >>= 1) sq += __shfl_xor(sq, off, 32);
        if (lane == 0) s_red[wid] = sq;
        __syncthreads();
        if (tid == 0) {
            float tot = 0.0f;
            for (int i = 0; i < 16; ++i) tot += s_red[i];
            s_sc[0] = (tot * (1.0f / (float)CD) > 0.5f) ? 1.0f : 0.0f;
        }
        __syncthreads();
        float wmask = s_sc[0];

        // K/V write: K = K*(1-wa) + wa*kw  (wa = wmask * addr)
        for (int i = tid; i < 32768; i += 512) {
            int hh = i >> 13;
            int ss = (i >> 7) & 63;
            int dh = i & 127;
            float wa = wmask * s_addr[hh * 64 + ss];
            float om = 1.0f - wa;
            Km[i] = Km[i] * om + wa * s_k[hh * 128 + dh];
            Vm[i] = Vm[i] * om + wa * s_v[hh * 128 + dh];
        }

        // gate GEMV over [h, mem_out] and fuse
        float gz = gb[tid];
        for (int j = 0; j < CD; ++j)
            gz += s_h[j] * gW[j * CD + tid] + s_mo[j] * gW[(CD + j) * CD + tid];
        float gate = sigmoidf_(gz);
        float fpre = gate * s_h[tid] + (1.0f - gate) * s_mo[tid];

        // LayerNorm over 512 (two block reductions)
        float sm = fpre;
        for (int off = 16; off >= 1; off >>= 1) sm += __shfl_xor(sm, off, 32);
        if (lane == 0) s_red[wid] = sm;
        __syncthreads();
        if (tid == 0) {
            float tot = 0.0f;
            for (int i = 0; i < 16; ++i) tot += s_red[i];
            s_sc[0] = tot * (1.0f / (float)CD);
        }
        __syncthreads();
        float mean = s_sc[0];
        float dv = fpre - mean;
        float vv = dv * dv;
        for (int off = 16; off >= 1; off >>= 1) vv += __shfl_xor(vv, off, 32);
        if (lane == 0) s_red[wid] = vv;
        __syncthreads();
        if (tid == 0) {
            float tot = 0.0f;
            for (int i = 0; i < 16; ++i) tot += s_red[i];
            s_sc[1] = rsqrtf(tot * (1.0f / (float)CD) + 1e-5f);
        }
        __syncthreads();
        float inv = s_sc[1];
        h[row + tid] = dv * inv * lnw[tid] + lnb[tid];
        __syncthreads();   // protect s_h / s_attn / s_addr / K,V for next step
    }
}

// ---------------------------------------------------------------------------
// Host orchestration
// ---------------------------------------------------------------------------
extern "C" void kernel_launch(void* const* d_in, const int* in_sizes, int n_in,
                              void* d_out, int out_size, void* d_ws, size_t ws_size,
                              hipStream_t stream)
{
    (void)in_sizes; (void)n_in; (void)out_size; (void)ws_size;

    const int* tokens      = (const int*)  d_in[0];
    const float* embed     = (const float*)d_in[1];
    const float* blk_ln_w  = (const float*)d_in[2];
    const float* blk_ln_b  = (const float*)d_in[3];
    const float* blk_inprj = (const float*)d_in[4];
    const float* blk_cw    = (const float*)d_in[5];
    const float* blk_cb    = (const float*)d_in[6];
    const float* blk_xprj  = (const float*)d_in[7];
    const float* blk_dtw   = (const float*)d_in[8];
    const float* blk_dtb   = (const float*)d_in[9];
    const float* blk_Alog  = (const float*)d_in[10];
    const float* blk_D     = (const float*)d_in[11];
    const float* blk_oprj  = (const float*)d_in[12];
    const float* Wq        = (const float*)d_in[13];
    const float* Wk        = (const float*)d_in[14];
    const float* Wv        = (const float*)d_in[15];
    const float* Wo        = (const float*)d_in[16];
    const float* gate_W    = (const float*)d_in[17];
    const float* gate_b    = (const float*)d_in[18];
    const float* mem_ln_w  = (const float*)d_in[19];
    const float* mem_ln_b  = (const float*)d_in[20];
    const float* lnf_w     = (const float*)d_in[21];
    const float* lnf_b     = (const float*)d_in[22];
    const float* lm_head   = (const float*)d_in[23];
    float* out             = (float*)d_out;

    // Workspace layout (floats). y is written in-place over xc; proj reuses bufA.
    float* h    = (float*)d_ws;                        // BT*D
    float* bufA = h    + (size_t)CBT * CD;             // BT*D   (x_ln / proj / final-LN)
    float* xz   = bufA + (size_t)CBT * CD;             // BT*2*DI
    float* xc   = xz   + (size_t)CBT * 2 * CDI;        // BT*DI
    float* dtb  = xc   + (size_t)CBT * CDI;            // BT*DI
    float* proj = bufA;                                // BT*64 (fits inside bufA)

    // Embedding
    {
        int total = CBT * CD;
        embed_kernel<<<(total + 255) / 256, 256, 0, stream>>>(tokens, embed, h, total);
    }

    // Mamba layers
    for (int l = 0; l < CL; ++l) {
        // LN(h) -> bufA
        ln_kernel<<<CBT / 4, 128, 0, stream>>>(h, blk_ln_w + l * CD, blk_ln_b + l * CD, bufA, CBT);

        // xz = x @ in_proj   (16384x512 @ 512x2048)
        gemm_wmma<0, false, false><<<dim3(2 * CDI / 64, CBT / 64), 256, 0, stream>>>(
            bufA, blk_inprj + (size_t)l * CD * 2 * CDI, nullptr, xz,
            CBT, 2 * CDI, CD, CD, 2 * CDI, 2 * CDI);

        // causal conv + SiLU -> xc
        {
            int total = CBT * CDI;
            conv_silu_kernel<<<(total + 255) / 256, 256, 0, stream>>>(
                xz, blk_cw + (size_t)l * CDI * 4, blk_cb + (size_t)l * CDI, xc, total);
        }

        // proj = xc @ x_proj   (16384x1024 @ 1024x64)
        gemm_wmma<0, false, false><<<dim3(1, CBT / 64), 256, 0, stream>>>(
            xc, blk_xprj + (size_t)l * CDI * 64, nullptr, proj,
            CBT, 64, CDI, CDI, 64, 64);

        // dt = softplus(dt_r @ dt_w + dt_b)   (16384x32 @ 32x1024, A strided in proj)
        gemm_wmma<1, true, false><<<dim3(CDI / 64, CBT / 64), 256, 0, stream>>>(
            proj, blk_dtw + (size_t)l * 32 * CDI, blk_dtb + (size_t)l * CDI, dtb,
            CBT, CDI, 32, 64, CDI, CDI);

        // selective scan: y = (scan + D*x) * silu(z), in-place into xc
        scan_kernel<<<32, 256, 0, stream>>>(
            dtb, proj, xz, blk_Alog + (size_t)l * CDI * 16, blk_D + (size_t)l * CDI, xc);

        // h += y @ out_proj   (16384x1024 @ 1024x512, residual)
        gemm_wmma<0, false, true><<<dim3(CD / 64, CBT / 64), 256, 0, stream>>>(
            xc, blk_oprj + (size_t)l * CDI * CD, nullptr, h,
            CBT, CD, CDI, CDI, CD, CD);
    }

    // Memory controller: 8 persistent WGs, 256 KB dynamic LDS of K/V per WG
    {
        size_t shmem = 2 * 4 * 64 * 128 * sizeof(float);
        memctrl_kernel<<<CB, 512, shmem, stream>>>(
            h, Wq, Wk, Wv, Wo, gate_W, gate_b, mem_ln_w, mem_ln_b);
    }

    // Final LN + LM head
    ln_kernel<<<CBT / 4, 128, 0, stream>>>(h, lnf_w, lnf_b, bufA, CBT);
    gemm_wmma<0, false, false><<<dim3(CV / 64, CBT / 64), 256, 0, stream>>>(
        bufA, lm_head, nullptr, out, CBT, CV, CD, CD, CV, CV);
}
